// MultiHeadSelfAttention_618475291166
// MI455X (gfx1250) — matrix operand order repaired and checked by reading, not compiled, not run on hardware
//
#include <hip/hip_runtime.h>

// ---------------------------------------------------------------------------
// MultiHeadSelfAttention for MI455X (gfx1250): TDM -> LDS -> bf16 WMMA.
// B=4, N=2048, C=1024, H=16, D=64.
// ---------------------------------------------------------------------------

#define BATCH     4
#define SEQ       2048
#define EMBED     1024
#define HEADS     16
#define HDIM      64
#define QKV_COLS  (3 * EMBED)
#define ROWS      (BATCH * SEQ)          // 8192
#define SCALE_F   0.125f                 // 64^-0.5

typedef __bf16 v16bf __attribute__((ext_vector_type(16)));
typedef float  v8f   __attribute__((ext_vector_type(8)));
typedef unsigned int u32x4 __attribute__((ext_vector_type(4)));
typedef int          i32x4 __attribute__((ext_vector_type(4)));
typedef int          i32x8 __attribute__((ext_vector_type(8)));

union Frag16 { v16bf v; u32x4 q[2]; };

// round-to-nearest-even f32 -> bf16 bits
__device__ __forceinline__ unsigned short f2bf(float f) {
    union { float f; unsigned u; } c; c.f = f;
    unsigned r = c.u + 0x7FFFu + ((c.u >> 16) & 1u);
    return (unsigned short)(r >> 16);
}

__device__ __forceinline__ v8f wmma_bf16(v16bf a, v16bf b, v8f c) {
    return __builtin_amdgcn_wmma_f32_16x16x32_bf16(false, a, false, b,
                                                   (short)0, c, false, false);
}

// A-fragment: 16 contiguous halves at p
__device__ __forceinline__ v16bf frag_a_ptr(const unsigned short* p) {
    Frag16 f;
    f.q[0] = *(const u32x4*)(p);
    f.q[1] = *(const u32x4*)(p + 8);
    return f.v;
}
// B-fragment: 16 contiguous halves at p
__device__ __forceinline__ v16bf frag_b_ptr(const unsigned short* p) {
    Frag16 f;
    f.q[0] = *(const u32x4*)(p);
    f.q[1] = *(const u32x4*)(p + 8);
    return f.v;
}

// A-fragment (16x32 bf16) direct from row-major global [ld], tile (row0,k0)
__device__ __forceinline__ v16bf load_a_frag_g(const unsigned short* __restrict__ base,
                                               int ld, int row0, int k0, int lane) {
    int m  = lane & 15;
    int kk = k0 + ((lane >> 4) << 4);
    return frag_a_ptr(base + (size_t)(row0 + m) * ld + kk);
}

// LDS byte offset of a shared-memory pointer (generic addr low 32 bits)
__device__ __forceinline__ unsigned lds_off(const void* p) {
    return (unsigned)(unsigned long long)p;
}

// ---------------------------------------------------------------------------
// Tensor Data Mover: 2D bf16 tile load, global -> LDS.
// tile_d0 = contiguous elements per row, tile_d1 = rows,
// stride_elems = row pitch of the global tensor (elements).
// ---------------------------------------------------------------------------
__device__ __forceinline__ void tdm_load_2d(const void* lds_dst, const void* gsrc,
                                            unsigned tile_d0, unsigned tile_d1,
                                            unsigned stride_elems) {
    unsigned long long ga = (unsigned long long)gsrc;
    u32x4 g0;
    g0[0] = 1u;                                            // count=1 (valid), user D#
    g0[1] = lds_off(lds_dst);                              // lds_addr
    g0[2] = (unsigned)ga;                                  // global_addr[31:0]
    g0[3] = (unsigned)((ga >> 32) & 0x01FFFFFFu)           // global_addr[56:32]
          | 0x80000000u;                                   // type=2 ("image")
    i32x8 g1;
    g1[0] = 0x00010000;                                    // data_size=1 -> 2 bytes
    g1[1] = 0;                                             // tensor_dim0[15:0]=0 (dim=2^30)
    g1[2] = 0x4000;                                        // td0[31:16]=2^14, td1 lo=0
    g1[3] = 0x4000 | (int)(tile_d0 << 16);                 // td1[31:16]=2^14, tile_dim0
    g1[4] = (int)(tile_d1 & 0xFFFFu);                      // tile_dim1 (tile_dim2=0)
    g1[5] = (int)stride_elems;                             // tensor_dim0_stride lo32
    g1[6] = 0;
    g1[7] = 0;
    i32x4 gz = {0, 0, 0, 0};
#if defined(__clang_major__) && (__clang_major__ >= 23)
    i32x8 gz8 = {0, 0, 0, 0, 0, 0, 0, 0};
    __builtin_amdgcn_tensor_load_to_lds(g0, g1, gz, gz, gz8, 0);
#else
    __builtin_amdgcn_tensor_load_to_lds(g0, g1, gz, gz, 0);
#endif
}

// ---------------------------------------------------------------------------
// Stage 0a: f32 -> bf16 elementwise (for x)
// ---------------------------------------------------------------------------
__global__ void cvt_bf16_kernel(const float* __restrict__ in,
                                unsigned short* __restrict__ out, int n) {
    for (int i = blockIdx.x * blockDim.x + threadIdx.x; i < n;
         i += gridDim.x * blockDim.x)
        out[i] = f2bf(in[i]);
}

// Stage 0b: transpose + convert: Wt[n*K + k] = bf16(W[k*N + n])
__global__ void transpose_bf16_kernel(const float* __restrict__ W,
                                      unsigned short* __restrict__ Wt,
                                      int K, int N) {
    int total = K * N;
    for (int i = blockIdx.x * blockDim.x + threadIdx.x; i < total;
         i += gridDim.x * blockDim.x) {
        int n = i / K, k = i - n * K;
        Wt[(size_t)n * K + k] = f2bf(W[(size_t)k * N + n]);
    }
}

// ---------------------------------------------------------------------------
// Stage 1: QKV GEMM. Block = 8 waves = 128(M) x 128(N) tile, K-loop of 32.
// TDM double-buffers A(128x32) and B(128x32) tiles into LDS.
// Per wave / step: 1 A-fragment, 8 B-fragments, 8 WMMA.
// Epilogue adds bias, scatters bf16 to Q[b,h,n,d], K[b,h,n,d], Vt[b,h,d,n].
// ---------------------------------------------------------------------------
__global__ __launch_bounds__(256)
void gemm_qkv_kernel(const unsigned short* __restrict__ xb,
                     const unsigned short* __restrict__ wt,   // [3072][1024]
                     const float* __restrict__ bias,
                     unsigned short* __restrict__ Qb,
                     unsigned short* __restrict__ Kb,
                     unsigned short* __restrict__ Vt) {
    __shared__ unsigned short Abuf[2][128 * 32];
    __shared__ unsigned short Bbuf[2][128 * 32];
    const int NTB = QKV_COLS / 128;                      // 24 col-blocks
    int mblk = blockIdx.x / NTB;
    int cblk = blockIdx.x - mblk * NTB;
    int Mbase = mblk * 128, c0 = cblk * 128;
    int tid = threadIdx.x, w = tid >> 5, lane = tid & 31;
    const int NSTEP = EMBED / 32;                        // 32

    if (tid < 32) {
        tdm_load_2d(Abuf[0], xb + (size_t)Mbase * EMBED, 32, 128, EMBED);
        tdm_load_2d(Bbuf[0], wt + (size_t)c0 * EMBED, 32, 128, EMBED);
    }

    v8f acc[8] = {};
    for (int s = 0; s < NSTEP; ++s) {
        int cur = s & 1;
        if (tid < 32) {
            if (s + 1 < NSTEP) {
                int k0 = (s + 1) * 32;
                tdm_load_2d(Abuf[cur ^ 1], xb + (size_t)Mbase * EMBED + k0, 32, 128, EMBED);
                tdm_load_2d(Bbuf[cur ^ 1], wt + (size_t)c0 * EMBED + k0, 32, 128, EMBED);
                __builtin_amdgcn_s_wait_tensorcnt(2);    // tile s complete
            } else {
                __builtin_amdgcn_s_wait_tensorcnt(0);
            }
        }
        __syncthreads();

        const unsigned short* ap =
            &Abuf[cur][(w * 16 + (lane & 15)) * 32 + ((lane >> 4) << 4)];
        v16bf a = frag_a_ptr(ap);
#pragma unroll
        for (int t = 0; t < 8; ++t) {
            const unsigned short* bp =
                &Bbuf[cur][(t * 16 + (lane & 15)) * 32 + ((lane >> 4) << 4)];
            acc[t] = wmma_bf16(a, frag_b_ptr(bp), acc[t]);
        }
        __syncthreads();
    }

    int n = lane & 15;
    int mofs = (lane >> 4) << 3;
    int row0 = Mbase + w * 16;
#pragma unroll
    for (int t = 0; t < 8; ++t) {
        int cc  = c0 + t * 16;               // 16-aligned, within one head
        int sel = cc / EMBED;
        int h   = (cc % EMBED) / HDIM;
        int d   = (cc % HDIM) + n;           // 0..63 within head
        float bv = bias[cc + n];
#pragma unroll
        for (int r = 0; r < 8; ++r) {
            int grow = row0 + r + mofs;
            int b  = grow >> 11;             // /2048
            int nq = grow & 2047;
            unsigned short val = f2bf(acc[t][r] + bv);
            size_t bh = (size_t)b * HEADS + h;
            if (sel == 0)      Qb[(bh * SEQ + nq) * HDIM + d] = val;
            else if (sel == 1) Kb[(bh * SEQ + nq) * HDIM + d] = val;
            else               Vt[(bh * HDIM + d) * SEQ + nq] = val;
        }
    }
}

// ---------------------------------------------------------------------------
// Stage 2: flash attention. Block = 8 waves sharing one (b,h); each wave owns
// one 16-query tile. K(32x64) and V(64x32) tiles TDM-staged per 32-key step.
// ---------------------------------------------------------------------------
__global__ __launch_bounds__(256)
void attn_kernel(const unsigned short* __restrict__ Qb,
                 const unsigned short* __restrict__ Kb,
                 const unsigned short* __restrict__ Vt,
                 unsigned short* __restrict__ Ob) {
    __shared__ unsigned short Ktile[2][32 * 64];   // [key][d]
    __shared__ unsigned short Vtile[2][64 * 32];   // [d][key]
    __shared__ unsigned short plds[8][16 * 40];    // per-wave P (16x32 + pad)

    int tid = threadIdx.x, w = tid >> 5, lane = tid & 31;
    int bh = blockIdx.x >> 4;                      // 0..63
    int qg = blockIdx.x & 15;
    int q0 = (qg * 8 + w) * 16;

    const unsigned short* Qp = Qb + (size_t)bh * SEQ * HDIM;
    const unsigned short* Kp = Kb + (size_t)bh * SEQ * HDIM;
    const unsigned short* Vp = Vt + (size_t)bh * HDIM * SEQ;
    unsigned short* myp = &plds[w][0];

    v16bf qf0 = load_a_frag_g(Qp, HDIM, q0, 0, lane);
    v16bf qf1 = load_a_frag_g(Qp, HDIM, q0, 32, lane);

    float mrow[8], lrow[8];
    v8f acc[4] = {};
#pragma unroll
    for (int r = 0; r < 8; ++r) { mrow[r] = -3.0e38f; lrow[r] = 0.0f; }

    int mofs = (lane >> 4) << 3;
    int ncol = lane & 15;
    const int NSTEP = SEQ / 32;                    // 64

    if (tid < 32) {
        tdm_load_2d(Ktile[0], Kp, 64, 32, 64);     // keys 0..31, all d
        tdm_load_2d(Vtile[0], Vp, 32, 64, SEQ);    // all d, keys 0..31
    }

    for (int s = 0; s < NSTEP; ++s) {
        int cur = s & 1;
        int n0 = s * 32;
        if (tid < 32) {
            if (s + 1 < NSTEP) {
                tdm_load_2d(Ktile[cur ^ 1], Kp + (size_t)(n0 + 32) * HDIM, 64, 32, 64);
                tdm_load_2d(Vtile[cur ^ 1], Vp + (n0 + 32), 32, 64, SEQ);
                __builtin_amdgcn_s_wait_tensorcnt(2);
            } else {
                __builtin_amdgcn_s_wait_tensorcnt(0);
            }
        }
        __syncthreads();

        // S tiles from LDS K: B-frag lane -> key = sub*16 + (lane&15)
        v8f s0 = {}, s1 = {};
        {
            const unsigned short* kb = &Ktile[cur][0];
            int krow0 = (ncol) * 64 + ((lane >> 4) << 4);
            int krow1 = (16 + ncol) * 64 + ((lane >> 4) << 4);
            s0 = wmma_bf16(qf0, frag_b_ptr(kb + krow0), s0);
            s0 = wmma_bf16(qf1, frag_b_ptr(kb + krow0 + 32), s0);
            s1 = wmma_bf16(qf0, frag_b_ptr(kb + krow1), s1);
            s1 = wmma_bf16(qf1, frag_b_ptr(kb + krow1 + 32), s1);
        }

#pragma unroll
        for (int r = 0; r < 8; ++r) {
            float a0 = s0[r] * SCALE_F, a1 = s1[r] * SCALE_F;
            float rm = fmaxf(a0, a1);
            rm = fmaxf(rm, __shfl_xor(rm, 1));
            rm = fmaxf(rm, __shfl_xor(rm, 2));
            rm = fmaxf(rm, __shfl_xor(rm, 4));
            rm = fmaxf(rm, __shfl_xor(rm, 8));
            float mnew = fmaxf(mrow[r], rm);
            float aold = __expf(mrow[r] - mnew);
            float p0 = __expf(a0 - mnew), p1 = __expf(a1 - mnew);
            float rs = p0 + p1;
            rs += __shfl_xor(rs, 1);
            rs += __shfl_xor(rs, 2);
            rs += __shfl_xor(rs, 4);
            rs += __shfl_xor(rs, 8);
            lrow[r] = lrow[r] * aold + rs;
            mrow[r] = mnew;
            acc[0][r] *= aold; acc[1][r] *= aold;
            acc[2][r] *= aold; acc[3][r] *= aold;
            int m = r + mofs;
            myp[m * 40 + ncol]      = f2bf(p0);
            myp[m * 40 + 16 + ncol] = f2bf(p1);
        }
        // wave-private LDS RAW: P stores feed the A-fragment gather below
        asm volatile("s_wait_dscnt 0" ::: "memory");

        Frag16 pf;
        {
            const unsigned short* p = myp + (lane & 15) * 40 + ((lane >> 4) << 4);
            pf.q[0] = *(const u32x4*)(p);
            pf.q[1] = *(const u32x4*)(p + 8);
        }
#pragma unroll
        for (int t = 0; t < 4; ++t) {
            const unsigned short* vp =
                &Vtile[cur][(t * 16 + ncol) * 32 + ((lane >> 4) << 4)];
            acc[t] = wmma_bf16(pf.v, frag_b_ptr(vp), acc[t]);
        }
        __syncthreads();
    }

    // epilogue: normalize, store O[b, q, h*64 + d] as bf16 row-major [8192][1024]
    int b = bh >> 4, h = bh & 15;
    float linv[8];
#pragma unroll
    for (int r = 0; r < 8; ++r) linv[r] = 1.0f / lrow[r];
#pragma unroll
    for (int t = 0; t < 4; ++t) {
        int col = h * HDIM + t * 16 + ncol;
#pragma unroll
        for (int r = 0; r < 8; ++r) {
            int q = q0 + r + mofs;
            Ob[((size_t)(b * SEQ + q)) * EMBED + col] = f2bf(acc[t][r] * linv[r]);
        }
    }
}

// ---------------------------------------------------------------------------
// Stage 3: projection GEMM (TDM-staged, 128x128 block tile), fp32 output.
// ---------------------------------------------------------------------------
__global__ __launch_bounds__(256)
void gemm_proj_kernel(const unsigned short* __restrict__ Ob,
                      const unsigned short* __restrict__ wt,   // [1024][1024]
                      const float* __restrict__ bias,
                      float* __restrict__ out) {
    __shared__ unsigned short Abuf[2][128 * 32];
    __shared__ unsigned short Bbuf[2][128 * 32];
    const int NTB = EMBED / 128;                         // 8 col-blocks
    int mblk = blockIdx.x / NTB;
    int cblk = blockIdx.x - mblk * NTB;
    int Mbase = mblk * 128, c0 = cblk * 128;
    int tid = threadIdx.x, w = tid >> 5, lane = tid & 31;
    const int NSTEP = EMBED / 32;

    if (tid < 32) {
        tdm_load_2d(Abuf[0], Ob + (size_t)Mbase * EMBED, 32, 128, EMBED);
        tdm_load_2d(Bbuf[0], wt + (size_t)c0 * EMBED, 32, 128, EMBED);
    }

    v8f acc[8] = {};
    for (int s = 0; s < NSTEP; ++s) {
        int cur = s & 1;
        if (tid < 32) {
            if (s + 1 < NSTEP) {
                int k0 = (s + 1) * 32;
                tdm_load_2d(Abuf[cur ^ 1], Ob + (size_t)Mbase * EMBED + k0, 32, 128, EMBED);
                tdm_load_2d(Bbuf[cur ^ 1], wt + (size_t)c0 * EMBED + k0, 32, 128, EMBED);
                __builtin_amdgcn_s_wait_tensorcnt(2);
            } else {
                __builtin_amdgcn_s_wait_tensorcnt(0);
            }
        }
        __syncthreads();

        const unsigned short* ap =
            &Abuf[cur][(w * 16 + (lane & 15)) * 32 + ((lane >> 4) << 4)];
        v16bf a = frag_a_ptr(ap);
#pragma unroll
        for (int t = 0; t < 8; ++t) {
            const unsigned short* bp =
                &Bbuf[cur][(t * 16 + (lane & 15)) * 32 + ((lane >> 4) << 4)];
            acc[t] = wmma_bf16(a, frag_b_ptr(bp), acc[t]);
        }
        __syncthreads();
    }

    int n = lane & 15;
    int mofs = (lane >> 4) << 3;
    int row0 = Mbase + w * 16;
#pragma unroll
    for (int t = 0; t < 8; ++t) {
        int c = c0 + t * 16 + n;
        float bv = bias[c];
#pragma unroll
        for (int r = 0; r < 8; ++r) {
            int grow = row0 + r + mofs;
            out[(size_t)grow * EMBED + c] = acc[t][r] + bv;
        }
    }
}

// ---------------------------------------------------------------------------
// Host launcher
// ---------------------------------------------------------------------------
extern "C" void kernel_launch(void* const* d_in, const int* in_sizes, int n_in,
                              void* d_out, int out_size, void* d_ws, size_t ws_size,
                              hipStream_t stream) {
    const float* x     = (const float*)d_in[0];
    const float* Wqkv  = (const float*)d_in[1];
    const float* bqkv  = (const float*)d_in[2];
    const float* Wproj = (const float*)d_in[3];
    const float* bproj = (const float*)d_in[4];
    float* out = (float*)d_out;

    char* ws = (char*)d_ws;
    size_t off = 0;
    unsigned short* xb     = (unsigned short*)(ws + off); off += (size_t)ROWS * EMBED * 2;
    unsigned short* WqkvT  = (unsigned short*)(ws + off); off += (size_t)QKV_COLS * EMBED * 2;
    unsigned short* WprojT = (unsigned short*)(ws + off); off += (size_t)EMBED * EMBED * 2;
    unsigned short* Qb     = (unsigned short*)(ws + off); off += (size_t)ROWS * EMBED * 2;
    unsigned short* Kb     = (unsigned short*)(ws + off); off += (size_t)ROWS * EMBED * 2;
    unsigned short* Vt     = (unsigned short*)(ws + off); off += (size_t)ROWS * EMBED * 2;
    unsigned short* Ob     = (unsigned short*)(ws + off); off += (size_t)ROWS * EMBED * 2;

    // Stage 0: precision conversion / weight transposes
    cvt_bf16_kernel<<<4096, 256, 0, stream>>>(x, xb, ROWS * EMBED);
    transpose_bf16_kernel<<<6144, 256, 0, stream>>>(Wqkv, WqkvT, EMBED, QKV_COLS);
    transpose_bf16_kernel<<<2048, 256, 0, stream>>>(Wproj, WprojT, EMBED, EMBED);

    // Stage 1: QKV GEMM  (64 m-blocks * 24 col-blocks)
    gemm_qkv_kernel<<<1536, 256, 0, stream>>>(xb, WqkvT, bqkv, Qb, Kb, Vt);

    // Stage 2: attention (64 bh * 16 q-groups)
    attn_kernel<<<1024, 256, 0, stream>>>(Qb, Kb, Vt, Ob);

    // Stage 3: projection (64 m-blocks * 8 col-blocks)
    gemm_proj_kernel<<<512, 256, 0, stream>>>(Ob, WprojT, bproj, out);
}
